// FastSpeech_60687887893013
// MI455X (gfx1250) — matrix operand order repaired and checked by reading, not compiled, not run on hardware
//
#include <hip/hip_runtime.h>

#define BB 8
#define LL 128
#define TT 1024
#define DD 256
#define HH 2
#define DKK 128
#define DII 1024
#define NLAYERS 4
#define NMEL 80

typedef __attribute__((ext_vector_type(16))) __bf16 v16bf;
typedef __attribute__((ext_vector_type(8)))  float  v8f;

__device__ __forceinline__ unsigned short f2bf(float x) {
  unsigned u = __float_as_uint(x);
  u += 0x7FFFu + ((u >> 16) & 1u);   // round-to-nearest-even
  return (unsigned short)(u >> 16);
}
__device__ __forceinline__ unsigned pack2bf(float a, float b) {
  return (unsigned)f2bf(a) | ((unsigned)f2bf(b) << 16);
}

// ---------------------------------------------------------------------------
// Templated WMMA GEMM:  C[M,N] = act( A' @ W' + bias )
//   bf16 operands (converted on the fly), fp32 accumulate via
//   v_wmma_f32_16x16x32_bf16.
//   Block tile 128(M) x 64(N), stage depth BK=64, 8 waves/block.
//   Wave grid 4(M) x 2(N): each wave owns 32x32 output =
//   2 A-frags x 2 B-frags x 4 accumulators -> 8 WMMAs per LDS stage.
// AMODE 0: A row-major MxK
// AMODE 1: implicit im2col of x:(BB,Tlen,Cin); K = Cin*ks; zero-pad `pad`
// WMODE 0: W row-major KxN
// WMODE 1: W[n*K + k]   (covers (Cout,Cin,ks) conv weight layout)
// ---------------------------------------------------------------------------
template <int AMODE, int WMODE, int RELU>
__global__ __launch_bounds__(256) void k_gemm(
    const float* __restrict__ A, const float* __restrict__ W,
    const float* __restrict__ bias, float* __restrict__ C,
    int M, int N, int K, int Tlen, int Cin, int ks, int pad) {
  __shared__ unsigned As[128][33];   // 128 rows x 32 K-pairs (bf16x2), padded
  __shared__ unsigned Bs[32][65];    // 32 K-pairs x 64 cols  (bf16x2), padded
  const int tid  = threadIdx.x;
  const int bm   = blockIdx.x * 128;
  const int bn   = blockIdx.y * 64;
  const int wave = tid >> 5, lane = tid & 31;
  const int wm   = wave & 3,  wn   = wave >> 2;   // 4 x 2 wave grid
  const int lm   = lane & 15, lh   = lane >> 4;

  v8f acc[4];
#pragma unroll
  for (int i = 0; i < 4; ++i) acc[i] = (v8f){0.f,0.f,0.f,0.f,0.f,0.f,0.f,0.f};

  for (int k0 = 0; k0 < K; k0 += 64) {
    __syncthreads();
    // L2 warm-up for the next W tile (emits global_prefetch_b8)
    if (k0 + 64 < K && tid < 64) {
      int col = bn + tid;
      if (col < N) {
        const float* nw = (WMODE == 0) ? &W[(size_t)(k0 + 64) * N + col]
                                       : &W[(size_t)col * K + (k0 + 64)];
        __builtin_prefetch(nw, 0, 1);
      }
    }
    // --- stage A tile: 128 x 32 packed bf16 pairs, 16 entries/thread ---
#pragma unroll
    for (int e = tid; e < 128 * 32; e += 256) {
      int i = e >> 5, j = e & 31;
      int m = bm + i, kk = k0 + 2 * j;
      float a0, a1;
      if (AMODE == 0) {
        const float* ap = A + (size_t)m * K + kk;
        a0 = ap[0]; a1 = ap[1];
      } else {
        int b  = m / Tlen, t = m - b * Tlen;
        int c0 = kk / ks,       d0 = kk - c0 * ks;
        int c1 = (kk + 1) / ks, d1 = (kk + 1) - c1 * ks;
        int t0 = t + d0 - pad,  t1 = t + d1 - pad;
        a0 = (t0 >= 0 && t0 < Tlen) ? A[((size_t)b * Tlen + t0) * Cin + c0] : 0.f;
        a1 = (t1 >= 0 && t1 < Tlen) ? A[((size_t)b * Tlen + t1) * Cin + c1] : 0.f;
      }
      As[i][j] = pack2bf(a0, a1);
    }
    // --- stage B tile: 32 x 64 packed bf16 pairs, 8 entries/thread ---
#pragma unroll
    for (int e = tid; e < 32 * 64; e += 256) {
      int j = e >> 6, n = e & 63;
      int col = bn + n, kk = k0 + 2 * j;
      float b0 = 0.f, b1 = 0.f;
      if (col < N) {
        if (WMODE == 0) { b0 = W[(size_t)kk * N + col];  b1 = W[(size_t)(kk + 1) * N + col]; }
        else            { b0 = W[(size_t)col * K + kk];  b1 = W[(size_t)col * K + kk + 1];  }
      }
      Bs[j][n] = pack2bf(b0, b1);
    }
    __syncthreads();

    // --- two 32-deep WMMA half-steps per stage ---
#pragma unroll
    for (int s = 0; s < 2; ++s) {
      const int jb = s * 16;
      union { v16bf v; unsigned u[8]; } fa0, fa1, fb0, fb1;
#pragma unroll
      for (int r = 0; r < 8; ++r) {           // A frag: lane = row l%16
        int j = jb + (r >> 2) * 8 + lh * 4 + (r & 3);
        fa0.u[r] = As[wm * 32 + lm][j];
        fa1.u[r] = As[wm * 32 + 16 + lm][j];
      }
#pragma unroll
      for (int r = 0; r < 8; ++r) {           // B frag: lane = col, VGPR = K-pair
        int j = jb + (r >> 2) * 8 + lh * 4 + (r & 3);
        fb0.u[r] = Bs[j][wn * 32 + lm];
        fb1.u[r] = Bs[j][wn * 32 + 16 + lm];
      }
      acc[0] = __builtin_amdgcn_wmma_f32_16x16x32_bf16(false, fa0.v, false, fb0.v,
                                                       (short)0, acc[0], false, false);
      acc[1] = __builtin_amdgcn_wmma_f32_16x16x32_bf16(false, fa0.v, false, fb1.v,
                                                       (short)0, acc[1], false, false);
      acc[2] = __builtin_amdgcn_wmma_f32_16x16x32_bf16(false, fa1.v, false, fb0.v,
                                                       (short)0, acc[2], false, false);
      acc[3] = __builtin_amdgcn_wmma_f32_16x16x32_bf16(false, fa1.v, false, fb1.v,
                                                       (short)0, acc[3], false, false);
    }
  }

  // --- epilogue: bias [+relu]; C/D layout row = v + 8*half, col = lane%16 ---
#pragma unroll
  for (int sub = 0; sub < 2; ++sub) {
#pragma unroll
    for (int tcol = 0; tcol < 2; ++tcol) {
      const v8f a = acc[sub * 2 + tcol];
      const int row0 = bm + wm * 32 + sub * 16 + lh * 8;
      const int col  = bn + wn * 32 + tcol * 16 + lm;
      if (col < N) {
        const float bv = bias[col];
#pragma unroll
        for (int r = 0; r < 8; ++r) {
          float val = a[r] + bv;
          if (RELU) val = fmaxf(val, 0.f);
          C[(size_t)(row0 + r) * N + col] = val;
        }
      }
    }
  }
}

// ---------------------------------------------------------------------------
// x[m,:] = tok_emb[seq[m]] + pos_emb[pos[m]];  np[m] = (seq!=0)
// ---------------------------------------------------------------------------
__global__ void k_embed(const int* __restrict__ seq, const int* __restrict__ pos,
                        const float* __restrict__ tok, const float* __restrict__ pe,
                        float* __restrict__ x, float* __restrict__ np) {
  int m = blockIdx.x, d = threadIdx.x;
  int s = seq[m], p = pos[m];
  x[(size_t)m * DD + d] = tok[(size_t)s * DD + d] + pe[(size_t)p * DD + d];
  if (d == 0) np[m] = (s != 0) ? 1.f : 0.f;
}

// ---------------------------------------------------------------------------
// y = [relu]( LN(a [+ res]) ) [* np] ; one 256-thread block per row (D=256)
// ---------------------------------------------------------------------------
__global__ __launch_bounds__(256) void k_layernorm(
    const float* __restrict__ a, const float* __restrict__ res,
    const float* __restrict__ g, const float* __restrict__ bb,
    const float* __restrict__ np, float* __restrict__ y, int relu) {
  __shared__ float red[8];
  int m = blockIdx.x, d = threadIdx.x;
  float v = a[(size_t)m * DD + d] + (res ? res[(size_t)m * DD + d] : 0.f);
  float s = v;
  for (int off = 16; off; off >>= 1) s += __shfl_down(s, off, 32);
  if ((d & 31) == 0) red[d >> 5] = s;
  __syncthreads();
  if (d == 0) { float t = 0.f; for (int i = 0; i < 8; ++i) t += red[i]; red[0] = t; }
  __syncthreads();
  float mean = red[0] * (1.f / DD);
  __syncthreads();
  float dv = v - mean;
  float s2 = dv * dv;
  for (int off = 16; off; off >>= 1) s2 += __shfl_down(s2, off, 32);
  if ((d & 31) == 0) red[d >> 5] = s2;
  __syncthreads();
  if (d == 0) { float t = 0.f; for (int i = 0; i < 8; ++i) t += red[i]; red[0] = t; }
  __syncthreads();
  float var = red[0] * (1.f / DD);
  float out = dv * rsqrtf(var + 1e-5f) * g[d] + bb[d];
  if (relu) out = fmaxf(out, 0.f);
  if (np) out *= np[m];
  y[(size_t)m * DD + d] = out;
}

// ---------------------------------------------------------------------------
// Unmasked softmax attention, one block per (lq, h, b); 128 threads.
// q/k/v/o layout: (B*L, H*DK)
// ---------------------------------------------------------------------------
__global__ __launch_bounds__(128) void k_attn(
    const float* __restrict__ q, const float* __restrict__ k,
    const float* __restrict__ v, float* __restrict__ o, int Lq, int Lk) {
  __shared__ float sc[TT];
  __shared__ float qs[DKK];
  __shared__ float red[4];
  int lq = blockIdx.x, h = blockIdx.y, b = blockIdx.z;
  int t = threadIdx.x;
  const size_t stride = (size_t)HH * DKK;
  qs[t] = q[((size_t)b * Lq + lq) * stride + h * DKK + t];
  __syncthreads();
  const float scale = 0.08838834764831845f;   // 1/sqrt(128)
  for (int m = t; m < Lk; m += 128) {
    const float* krow = k + ((size_t)b * Lk + m) * stride + h * DKK;
    float s = 0.f;
#pragma unroll 4
    for (int dd = 0; dd < DKK; ++dd) s += qs[dd] * krow[dd];
    sc[m] = s * scale;
  }
  __syncthreads();
  float mx = -1e30f;
  for (int m = t; m < Lk; m += 128) mx = fmaxf(mx, sc[m]);
  for (int off = 16; off; off >>= 1) mx = fmaxf(mx, __shfl_down(mx, off, 32));
  if ((t & 31) == 0) red[t >> 5] = mx;
  __syncthreads();
  if (t == 0) { float z = red[0]; for (int i = 1; i < 4; ++i) z = fmaxf(z, red[i]); red[0] = z; }
  __syncthreads();
  mx = red[0];
  __syncthreads();
  float sum = 0.f;
  for (int m = t; m < Lk; m += 128) { float e = __expf(sc[m] - mx); sc[m] = e; sum += e; }
  for (int off = 16; off; off >>= 1) sum += __shfl_down(sum, off, 32);
  if ((t & 31) == 0) red[t >> 5] = sum;
  __syncthreads();
  if (t == 0) { float z = 0.f; for (int i = 0; i < 4; ++i) z += red[i]; red[0] = z; }
  __syncthreads();
  float inv = 1.f / red[0];
  float acc = 0.f;
  for (int m = 0; m < Lk; ++m)
    acc += sc[m] * v[((size_t)b * Lk + m) * stride + h * DKK + t];
  o[((size_t)b * Lq + lq) * stride + h * DKK + t] = acc * inv;
}

// dur[m] = relu( h[m,:] . lw + lb ) * np[m]
__global__ __launch_bounds__(256) void k_durout(
    const float* __restrict__ h, const float* __restrict__ lw,
    const float* __restrict__ lb, const float* __restrict__ np,
    float* __restrict__ out) {
  __shared__ float red[8];
  int m = blockIdx.x, d = threadIdx.x;
  float s = h[(size_t)m * DD + d] * lw[d];
  for (int off = 16; off; off >>= 1) s += __shfl_down(s, off, 32);
  if ((d & 31) == 0) red[d >> 5] = s;
  __syncthreads();
  if (d == 0) {
    float t = 0.f; for (int i = 0; i < 8; ++i) t += red[i];
    out[m] = fmaxf(t + lb[0], 0.f) * np[m];
  }
}

__global__ void k_cumsum(const int* __restrict__ dur, int* __restrict__ cs) {
  int b = blockIdx.x;
  if (threadIdx.x == 0) {
    int c = 0;
    for (int j = 0; j < LL; ++j) { c += dur[b * LL + j]; cs[b * LL + j] = c; }
  }
}

// length regulator: xd[b,t,:] = enc[b, idx, :]*mask + pos_emb[mask? t+1 : 0]
__global__ __launch_bounds__(256) void k_expand(
    const float* __restrict__ enc, const int* __restrict__ cs,
    const float* __restrict__ pe, float* __restrict__ xd,
    float* __restrict__ npm) {
  int m = blockIdx.x;
  int b = m / TT, t = m - b * TT;
  int d = threadIdx.x;
  int total = cs[b * LL + LL - 1];
  int idx = 0;
  for (int j = 0; j < LL; ++j) idx += (cs[b * LL + j] <= t) ? 1 : 0;   // searchsorted 'right'
  if (idx > LL - 1) idx = LL - 1;
  int mask = (t < total) ? 1 : 0;
  int mp = mask ? (t + 1) : 0;
  float val = mask ? enc[((size_t)b * LL + idx) * DD + d] : 0.f;
  xd[(size_t)m * DD + d] = val + pe[(size_t)mp * DD + d];
  if (d == 0) npm[m] = (float)mask;
}

// ---------------------------------------------------------------------------
// Host orchestration
// ---------------------------------------------------------------------------
static void launch_gemm(const float* A, const float* W, const float* bias, float* C,
                        int M, int N, int K, int amode, int Tlen, int Cin, int ks,
                        int pad, int wmode, int relu, hipStream_t s) {
  dim3 grid(M / 128, (N + 63) / 64);
  if (amode == 0 && wmode == 0) {
    k_gemm<0, 0, 0><<<grid, 256, 0, s>>>(A, W, bias, C, M, N, K, Tlen, Cin, ks, pad);
  } else if (amode == 0 && wmode == 1) {
    k_gemm<0, 1, 0><<<grid, 256, 0, s>>>(A, W, bias, C, M, N, K, Tlen, Cin, ks, pad);
  } else if (relu) {
    k_gemm<1, 1, 1><<<grid, 256, 0, s>>>(A, W, bias, C, M, N, K, Tlen, Cin, ks, pad);
  } else {
    k_gemm<1, 1, 0><<<grid, 256, 0, s>>>(A, W, bias, C, M, N, K, Tlen, Cin, ks, pad);
  }
}

static void run_stack(void* const* d_in, int pi, float* x, float* q, float* k,
                      float* v, float* o, float* tmp, float* h, const float* np,
                      int Lcur, hipStream_t s) {
  const int M = BB * Lcur;
  const float* wq  = (const float*)d_in[pi + 0];
  const float* bq  = (const float*)d_in[pi + 1];
  const float* wk  = (const float*)d_in[pi + 2];
  const float* bk  = (const float*)d_in[pi + 3];
  const float* wv  = (const float*)d_in[pi + 4];
  const float* bv  = (const float*)d_in[pi + 5];
  const float* fc  = (const float*)d_in[pi + 6];
  const float* bfc = (const float*)d_in[pi + 7];
  const float* g1  = (const float*)d_in[pi + 8];
  const float* b1  = (const float*)d_in[pi + 9];
  const float* c1w = (const float*)d_in[pi + 10];
  const float* c1b = (const float*)d_in[pi + 11];
  const float* c2w = (const float*)d_in[pi + 12];
  const float* c2b = (const float*)d_in[pi + 13];
  const float* g2  = (const float*)d_in[pi + 14];
  const float* b2  = (const float*)d_in[pi + 15];
  const int HD = HH * DKK;
  for (int i = 0; i < NLAYERS; ++i) {
    size_t wO = (size_t)i * DD * HD;
    launch_gemm(x, wq + wO, bq + i * HD, q, M, HD, DD, 0, 0, 0, 0, 0, 0, 0, s);
    launch_gemm(x, wk + wO, bk + i * HD, k, M, HD, DD, 0, 0, 0, 0, 0, 0, 0, s);
    launch_gemm(x, wv + wO, bv + i * HD, v, M, HD, DD, 0, 0, 0, 0, 0, 0, 0, s);
    k_attn<<<dim3(Lcur, HH, BB), 128, 0, s>>>(q, k, v, o, Lcur, Lcur);
    launch_gemm(o, fc + (size_t)i * HD * DD, bfc + i * DD, tmp,
                M, DD, HD, 0, 0, 0, 0, 0, 0, 0, s);
    k_layernorm<<<M, DD, 0, s>>>(tmp, x, g1 + i * DD, b1 + i * DD, np, x, 0);
    launch_gemm(x, c1w + (size_t)i * DII * DD * 9, c1b + i * DII, h,
                M, DII, DD * 9, 1, Lcur, DD, 9, 4, 1, 1, s);      // conv k=9 + relu
    launch_gemm(h, c2w + (size_t)i * DD * DII, c2b + i * DD, tmp,
                M, DD, DII, 0, 0, 0, 0, 0, 1, 0, s);              // conv k=1
    k_layernorm<<<M, DD, 0, s>>>(tmp, x, g2 + i * DD, b2 + i * DD, np, x, 0);
  }
}

extern "C" void kernel_launch(void* const* d_in, const int* in_sizes, int n_in,
                              void* d_out, int out_size, void* d_ws, size_t ws_size,
                              hipStream_t stream) {
  (void)in_sizes; (void)n_in; (void)out_size; (void)ws_size;
  const int*   src_seq   = (const int*)d_in[0];
  const int*   src_pos   = (const int*)d_in[1];
  const int*   durations = (const int*)d_in[2];
  const float* tok_emb   = (const float*)d_in[4];
  const float* pos_emb   = (const float*)d_in[5];
  const float* mel_w     = (const float*)d_in[48];
  const float* mel_b     = (const float*)d_in[49];

  float* ws = (float*)d_ws;
  size_t off = 0;
  auto alloc = [&](size_t n) { float* p = ws + off; off += n; return p; };
  const size_t ML = (size_t)BB * LL, MT = (size_t)BB * TT;
  float* xe     = alloc(ML * DD);
  float* xd     = alloc(MT * DD);
  float* q      = alloc(MT * DD);
  float* k      = alloc(MT * DD);
  float* v      = alloc(MT * DD);
  float* o      = alloc(MT * DD);
  float* tmp    = alloc(MT * DD);
  float* h      = alloc(MT * DII);
  float* np_src = alloc(ML);
  float* np_mel = alloc(MT);
  float* h1     = alloc(ML * DD);
  float* h2     = alloc(ML * DD);
  int*   cs     = (int*)alloc(ML);

  float* mel_out = (float*)d_out;              // (B*T, 80)
  float* dur_out = (float*)d_out + MT * NMEL;  // (B*L,)

  // --- encoder ---
  k_embed<<<(int)ML, DD, 0, stream>>>(src_seq, src_pos, tok_emb, pos_emb, xe, np_src);
  run_stack(d_in, 6, xe, q, k, v, o, tmp, h, np_src, LL, stream);

  // --- duration predictor ---
  const float* dc1w = (const float*)d_in[38];
  const float* dc1b = (const float*)d_in[39];
  const float* dg1  = (const float*)d_in[40];
  const float* db1  = (const float*)d_in[41];
  const float* dc2w = (const float*)d_in[42];
  const float* dc2b = (const float*)d_in[43];
  const float* dg2  = (const float*)d_in[44];
  const float* db2  = (const float*)d_in[45];
  const float* dlw  = (const float*)d_in[46];
  const float* dlb  = (const float*)d_in[47];
  launch_gemm(xe, dc1w, dc1b, h1, (int)ML, DD, DD * 3, 1, LL, DD, 3, 1, 1, 0, stream);
  k_layernorm<<<(int)ML, DD, 0, stream>>>(h1, nullptr, dg1, db1, nullptr, h1, 1);
  launch_gemm(h1, dc2w, dc2b, h2, (int)ML, DD, DD * 3, 1, LL, DD, 3, 1, 1, 0, stream);
  k_layernorm<<<(int)ML, DD, 0, stream>>>(h2, nullptr, dg2, db2, nullptr, h2, 1);
  k_durout<<<(int)ML, DD, 0, stream>>>(h2, dlw, dlb, np_src, dur_out);

  // --- length regulator ---
  k_cumsum<<<BB, 32, 0, stream>>>(durations, cs);
  k_expand<<<(int)MT, DD, 0, stream>>>(xe, cs, pos_emb, xd, np_mel);

  // --- decoder ---
  run_stack(d_in, 22, xd, q, k, v, o, tmp, h, np_mel, TT, stream);

  // --- mel projection (N=80 = 5x16, column-bound checks handle tile edge) ---
  launch_gemm(xd, mel_w, mel_b, mel_out, (int)MT, NMEL, DD, 0, 0, 0, 0, 0, 0, 0, stream);
}
